// GPT2Attention_54528904790664
// MI455X (gfx1250) — matrix operand order repaired and checked by reading, not compiled, not run on hardware
//
#include <hip/hip_runtime.h>

// ---------------------------------------------------------------------------
// GPT2 attention block (B=4, S=1024, D=1024, H=16, dh=64), fp32 in/out.
// Strategy: bf16x3 split-precision WMMA (v_wmma_f32_16x16x32_bf16) for all
// matmuls; flash-style relu-softmax attention (no max/exp needed).
// GEMM uses double-buffered LDS with async global->LDS staging (ASYNCcnt).
// ---------------------------------------------------------------------------

typedef unsigned short u16;
typedef __attribute__((ext_vector_type(16))) __bf16 v16bf;
typedef __attribute__((ext_vector_type(8)))  float  v8f;
typedef __attribute__((ext_vector_type(4)))  unsigned int u32x4;
typedef __attribute__((ext_vector_type(2)))  unsigned int u32x2;

struct Frag16B2 { u32x4 q0, q1; };

__device__ inline v8f wmma_bf16f32(v16bf a, v16bf b, v8f c) {
  // D = A(16x32 bf16) * B(32x16 bf16) + C(16x16 f32)
  return __builtin_amdgcn_wmma_f32_16x16x32_bf16(
      /*neg_a=*/false, a, /*neg_b=*/false, b,
      /*c_mod=*/(short)0, c, /*reuse_a=*/false, /*reuse_b=*/false);
}

__device__ inline u16 f2bf_rne(float f) {
  unsigned u = __float_as_uint(f);
  unsigned r = u + 0x7FFFu + ((u >> 16) & 1u);
  return (u16)(r >> 16);
}

// Split fp32 x into bf16 hi (truncation -> exact residual) + bf16 lo (RNE).
__device__ inline void split2(float x, u16 &hi, u16 &lo) {
  unsigned u  = __float_as_uint(x);
  unsigned uh = u & 0xFFFF0000u;
  hi = (u16)(uh >> 16);
  lo = f2bf_rne(x - __uint_as_float(uh));
}

// A-matrix fragment (16x32 bf16), ISA layout:
//  lane r (0..15), half hf: row = r; v0-3 hold K = hf*8..hf*8+7,
//  v4-7 hold K = 16+hf*8..16+hf*8+7. Row-major LDS tile, stride in u16.
__device__ inline v16bf load_afrag(const u16* base, int stride, int r, int hf, int kOff) {
  const u16* p = base + r * stride + kOff + hf * 8;
  Frag16B2 f;
  f.q0 = *(const u32x4*)(p);
  f.q1 = *(const u32x4*)(p + 16);
  return __builtin_bit_cast(v16bf, f);
}

// B-matrix fragment (32x16 bf16), ISA layout:
//  lane r (0..15) holds column N=r; v0-3 hold K = hf*8..hf*8+7, v4-7 hold K = 16+hf*8..16+hf*8+7.
//  LDS tile stored column-major: tile[n][k], stride in u16.
__device__ inline v16bf load_bfrag(const u16* base, int stride, int r, int hf, int kOff) {
  const u16* p = base + r * stride + kOff + hf * 8;
  Frag16B2 f;
  f.q0 = *(const u32x4*)(p);
  f.q1 = *(const u32x4*)(p + 16);
  return __builtin_bit_cast(v16bf, f);
}

__device__ inline v8f zero8() { v8f z; for (int i = 0; i < 8; i++) z[i] = 0.0f; return z; }

// CDNA5 async global->LDS copy (16B per lane), tracked on ASYNCcnt.
__device__ inline void async_load_b128(unsigned lds_addr, const void* gaddr) {
  asm volatile("global_load_async_to_lds_b128 %0, %1, off"
               :: "v"(lds_addr), "v"(gaddr)
               : "memory");
}
__device__ inline void wait_asynccnt0() {
  asm volatile("s_wait_asynccnt 0" ::: "memory");
}

// ---------------------------------------------------------------------------
// Kernel: fp32 -> (bf16 hi, bf16 lo) split
// ---------------------------------------------------------------------------
__global__ void split_kernel(const float* __restrict__ src,
                             u16* __restrict__ hi, u16* __restrict__ lo, int n) {
  int i = blockIdx.x * 256 + threadIdx.x;
  if (i < n) {
    u16 h, l;
    split2(src[i], h, l);
    hi[i] = h;
    lo[i] = l;
  }
}

// ---------------------------------------------------------------------------
// Kernel: C[M,N] = A[M,K] @ B[K,N] + bias[N], bf16x3 split-precision WMMA.
// Block = 256 threads (8 waves), tile 128x128, K-step 32.
// Double-buffered LDS; A staged via global_load_async_to_lds_b128; B staged
// through registers (needs a transpose into [n][k]).
// ---------------------------------------------------------------------------
__global__ __launch_bounds__(256, 2)
void gemm_bf16x3(const u16* __restrict__ Ahi, const u16* __restrict__ Alo,
                 const u16* __restrict__ Bhi, const u16* __restrict__ Blo,
                 const float* __restrict__ bias, float* __restrict__ C,
                 int M, int N, int K) {
  __shared__ __align__(16) u16 sAh[2][128 * 32];
  __shared__ __align__(16) u16 sAl[2][128 * 32];
  __shared__ __align__(16) u16 sBh[2][128 * 32];   // transposed: [n][k]
  __shared__ __align__(16) u16 sBl[2][128 * 32];

  const int t = threadIdx.x;
  const int wv = t >> 5;
  const int lane = t & 31, hf = lane >> 4, r = lane & 15;
  const int mBase = blockIdx.y * 128;
  const int nBase = blockIdx.x * 128;

  // async-stage one 128x32 A tile (hi+lo) into buffer `buf`
  auto stageA = [&](int buf, int k0) {
    for (int i = 0; i < 2; i++) {
      int c = t + i * 256;               // 16B chunk id, 0..511
      int row = c >> 1;                  // 2 chunks per 64B row
      int o16 = (c & 1) * 8;             // u16 offset within row
      size_t g = (size_t)(mBase + row) * K + (k0 + o16);
      async_load_b128((unsigned)(size_t)&sAh[buf][row * 32 + o16], &Ahi[g]);
      async_load_b128((unsigned)(size_t)&sAl[buf][row * 32 + o16], &Alo[g]);
    }
  };
  // load a 32x128 B tile into registers (coalesced 8B quads)
  auto loadB = [&](int k0, u32x2* qh, u32x2* ql) {
    for (int i = 0; i < 4; i++) {
      int idx = t + i * 256;             // 0..1023
      int kr = idx >> 5;                 // 0..31
      int c4 = (idx & 31) * 4;           // 0..124
      size_t g = (size_t)(k0 + kr) * N + (nBase + c4);
      qh[i] = *(const u32x2*)&Bhi[g];
      ql[i] = *(const u32x2*)&Blo[g];
    }
  };
  // store registers transposed into [n][k] LDS buffer `buf`
  auto storeB = [&](int buf, const u32x2* qh, const u32x2* ql) {
    for (int i = 0; i < 4; i++) {
      int idx = t + i * 256;
      int kr = idx >> 5;
      int c4 = (idx & 31) * 4;
      sBh[buf][(c4 + 0) * 32 + kr] = (u16)(qh[i][0] & 0xFFFFu);
      sBh[buf][(c4 + 1) * 32 + kr] = (u16)(qh[i][0] >> 16);
      sBh[buf][(c4 + 2) * 32 + kr] = (u16)(qh[i][1] & 0xFFFFu);
      sBh[buf][(c4 + 3) * 32 + kr] = (u16)(qh[i][1] >> 16);
      sBl[buf][(c4 + 0) * 32 + kr] = (u16)(ql[i][0] & 0xFFFFu);
      sBl[buf][(c4 + 1) * 32 + kr] = (u16)(ql[i][0] >> 16);
      sBl[buf][(c4 + 2) * 32 + kr] = (u16)(ql[i][1] & 0xFFFFu);
      sBl[buf][(c4 + 3) * 32 + kr] = (u16)(ql[i][1] >> 16);
    }
  };

  v8f acc[8];
  for (int i = 0; i < 8; i++) acc[i] = zero8();

  // prologue: fill buffer 0
  u32x2 qh[4], ql[4];
  stageA(0, 0);
  loadB(0, qh, ql);
  storeB(0, qh, ql);
  wait_asynccnt0();
  __syncthreads();

  int cur = 0;
  for (int k0 = 0; k0 < K; k0 += 32) {
    const int nxt = cur ^ 1;
    const bool more = (k0 + 32 < K);
    // kick off next tile's traffic before computing (overlap with WMMAs)
    if (more) {
      stageA(nxt, k0 + 32);
      loadB(k0 + 32, qh, ql);
    }

    // compute from current buffer: 24 WMMAs
    v16bf aH = load_afrag(&sAh[cur][(wv * 16) * 32], 32, r, hf, 0);
    v16bf aL = load_afrag(&sAl[cur][(wv * 16) * 32], 32, r, hf, 0);
    for (int nt = 0; nt < 8; nt++) {
      v16bf bH = load_bfrag(&sBh[cur][(nt * 16) * 32], 32, r, hf, 0);
      v16bf bL = load_bfrag(&sBl[cur][(nt * 16) * 32], 32, r, hf, 0);
      acc[nt] = wmma_bf16f32(aH, bH, acc[nt]);  // hi*hi
      acc[nt] = wmma_bf16f32(aH, bL, acc[nt]);  // hi*lo
      acc[nt] = wmma_bf16f32(aL, bH, acc[nt]);  // lo*hi
    }

    if (more) {
      storeB(nxt, qh, ql);
      wait_asynccnt0();
    }
    __syncthreads();
    cur = nxt;
  }

  // epilogue: add bias, write fp32 C
  const int mRow = mBase + wv * 16 + hf * 8;
  for (int nt = 0; nt < 8; nt++) {
    int col = nBase + nt * 16 + r;
    float bv = bias[col];
    for (int j = 0; j < 8; j++) {
      C[(size_t)(mRow + j) * N + col] = acc[nt][j] + bv;
    }
  }
}

// ---------------------------------------------------------------------------
// Kernel: sumV[b,h,d] = sum_s V[b,h,s,d]   (V at qkv[...,2048 + h*64 + d])
// ---------------------------------------------------------------------------
__global__ void sumv_kernel(const float* __restrict__ qkv, float* __restrict__ sumV) {
  int bh = blockIdx.x;             // 0..63
  int b = bh >> 4, h = bh & 15;
  int d = threadIdx.x;             // 0..63
  const float* Vg = qkv + ((size_t)b * 1024) * 3072 + 2048 + h * 64 + d;
  float s = 0.0f;
  for (int i = 0; i < 1024; i++) s += Vg[(size_t)i * 3072];
  sumV[bh * 64 + d] = s;
}

// ---------------------------------------------------------------------------
// Kernel: causal relu-softmax attention, flash-style single pass.
// grid = (B*H, S/128); block = 256 (8 waves); wave w owns 16 q-rows.
// ctx written as bf16 hi/lo split (feeds the output-proj GEMM).
// ---------------------------------------------------------------------------
__global__ __launch_bounds__(256, 1)
void attn_kernel(const float* __restrict__ qkv, const float* __restrict__ sumV,
                 u16* __restrict__ ctxHi, u16* __restrict__ ctxLo) {
  __shared__ __align__(16) u16 sQh[128 * 64];
  __shared__ __align__(16) u16 sQl[128 * 64];
  __shared__ __align__(16) u16 sKh[128 * 64];        // [krow][dh]
  __shared__ __align__(16) u16 sKl[128 * 64];
  __shared__ __align__(16) u16 sVh[64 * 128];        // transposed [dh][krow]
  __shared__ __align__(16) u16 sVl[64 * 128];
  __shared__ __align__(16) u16 sRh[8][16 * 32];      // per-wave score scratch
  __shared__ __align__(16) u16 sRl[8][16 * 32];

  const int t = threadIdx.x;
  const int wv = t >> 5;
  const int lane = t & 31, hf = lane >> 4, r = lane & 15;
  const int bh = blockIdx.x;
  const int b = bh >> 4, h = bh & 15;
  const int qb = blockIdx.y;
  const int qBase = qb * 128;
  const size_t rowStride = 3072;
  const float* base = qkv + ((size_t)b * 1024) * rowStride + h * 64;

  // ---- stage Q tile (128x64) once, split to bf16 hi/lo ----
  for (int i = 0; i < 32; i++) {
    int idx = t + i * 256;             // 0..8191
    int row = idx >> 6, col = idx & 63;
    float x = base[(size_t)(qBase + row) * rowStride + col];
    u16 hi, lo;
    split2(x, hi, lo);
    sQh[row * 64 + col] = hi;
    sQl[row * 64 + col] = lo;
  }

  v8f acc[4];
  for (int i = 0; i < 4; i++) acc[i] = zero8();
  float den[8];
  for (int i = 0; i < 8; i++) den[i] = 0.0f;

  for (int kb = 0; kb <= qb; kb++) {     // causal: skip k-blocks above diagonal
    int kBase = kb * 128;
    __syncthreads();                     // protect previous tile reads
    // ---- stage K (row-major) and V (transposed) tiles ----
    for (int i = 0; i < 32; i++) {
      int idx = t + i * 256;
      int row = idx >> 6, col = idx & 63;
      u16 hi, lo;
      float kx = base[(size_t)(kBase + row) * rowStride + 1024 + col];
      split2(kx, hi, lo);
      sKh[row * 64 + col] = hi;
      sKl[row * 64 + col] = lo;
      float vx = base[(size_t)(kBase + row) * rowStride + 2048 + col];
      split2(vx, hi, lo);
      sVh[col * 128 + row] = hi;
      sVl[col * 128 + row] = lo;
    }
    __syncthreads();

    const int qrow0 = qBase + wv * 16 + hf * 8;
    for (int p = 0; p < 4; p++) {        // K chunks of 32 rows
      // two 16-wide score tiles -> one 16x32 R chunk in per-wave LDS scratch
      for (int sub = 0; sub < 2; sub++) {
        int kt = p * 2 + sub;
        v8f s = zero8();
        for (int dc = 0; dc < 2; dc++) { // dh = 64 -> 2 K-chunks of 32
          v16bf aH = load_afrag(&sQh[(wv * 16) * 64], 64, r, hf, dc * 32);
          v16bf aL = load_afrag(&sQl[(wv * 16) * 64], 64, r, hf, dc * 32);
          v16bf bH = load_bfrag(&sKh[(kt * 16) * 64], 64, r, hf, dc * 32);
          v16bf bL = load_bfrag(&sKl[(kt * 16) * 64], 64, r, hf, dc * 32);
          s = wmma_bf16f32(aH, bH, s);
          s = wmma_bf16f32(aH, bL, s);
          s = wmma_bf16f32(aL, bH, s);
        }
        int kcol = kBase + kt * 16 + r;
        for (int e = 0; e < 8; e++) {
          float sv = s[e] * 0.125f;                       // 1/sqrt(64)
          float rv = (kcol <= qrow0 + e && sv > 0.0f) ? sv : 0.0f;
          den[e] += rv;
          u16 hi, lo;
          split2(rv, hi, lo);
          sRh[wv][(e + hf * 8) * 32 + sub * 16 + r] = hi;
          sRl[wv][(e + hf * 8) * 32 + sub * 16 + r] = lo;
        }
      }
      // ctx += R(16x32) @ V(32x64)   (same-wave LDS RAW, no barrier needed)
      v16bf aH = load_afrag(&sRh[wv][0], 32, r, hf, 0);
      v16bf aL = load_afrag(&sRl[wv][0], 32, r, hf, 0);
      for (int dt = 0; dt < 4; dt++) {
        v16bf bH = load_bfrag(&sVh[(dt * 16) * 128], 128, r, hf, p * 32);
        v16bf bL = load_bfrag(&sVl[(dt * 16) * 128], 128, r, hf, p * 32);
        acc[dt] = wmma_bf16f32(aH, bH, acc[dt]);
        acc[dt] = wmma_bf16f32(aH, bL, acc[dt]);
        acc[dt] = wmma_bf16f32(aL, bH, acc[dt]);
      }
    }
  }

  // ---- reduce denominators across the 16 lanes of each half ----
  for (int e = 0; e < 8; e++) {
    float d = den[e];
    d += __shfl_xor(d, 1);
    d += __shfl_xor(d, 2);
    d += __shfl_xor(d, 4);
    d += __shfl_xor(d, 8);
    den[e] = d + 1e-12f;
  }

  // ---- epilogue: ctx = (acc + (eps/S) * sumV) / den, write bf16 hi/lo ----
  const float epsOverS = 1e-12f / 1024.0f;
  const int qrow0 = qBase + wv * 16 + hf * 8;
  for (int dt = 0; dt < 4; dt++) {
    int col = dt * 16 + r;
    float sv = sumV[bh * 64 + col];
    for (int e = 0; e < 8; e++) {
      float val = (acc[dt][e] + epsOverS * sv) / den[e];
      size_t o = ((size_t)b * 1024 + (qrow0 + e)) * 1024 + h * 64 + col;
      u16 hi, lo;
      split2(val, hi, lo);
      ctxHi[o] = hi;
      ctxLo[o] = lo;
    }
  }
}

// ---------------------------------------------------------------------------
// Host-side orchestration
// ---------------------------------------------------------------------------
extern "C" void kernel_launch(void* const* d_in, const int* in_sizes, int n_in,
                              void* d_out, int out_size, void* d_ws, size_t ws_size,
                              hipStream_t stream) {
  (void)in_sizes; (void)n_in; (void)out_size; (void)ws_size;
  const float* X    = (const float*)d_in[0];   // [4,1024,1024]
  const float* Wqkv = (const float*)d_in[1];   // [1024,3072]
  const float* bqkv = (const float*)d_in[2];   // [3072]
  const float* Wp   = (const float*)d_in[3];   // [1024,1024]
  const float* bp   = (const float*)d_in[4];   // [1024]
  float* out = (float*)d_out;                  // [4,1024,1024]

  char* ws = (char*)d_ws;
  size_t off = 0;
  auto carve = [&](size_t bytes) -> char* {
    char* p = ws + off;
    off += (bytes + 255) & ~(size_t)255;
    return p;
  };

  const size_t M = 4096, D = 1024, N3 = 3072;
  u16*   Xhi  = (u16*)carve(M * D * 2);
  u16*   Xlo  = (u16*)carve(M * D * 2);
  u16*   W1hi = (u16*)carve(D * N3 * 2);
  u16*   W1lo = (u16*)carve(D * N3 * 2);
  u16*   W2hi = (u16*)carve(D * D * 2);
  u16*   W2lo = (u16*)carve(D * D * 2);
  float* QKV  = (float*)carve(M * N3 * 4);
  float* sumV = (float*)carve(64 * 64 * 4);
  u16*   Chi  = (u16*)carve(M * D * 2);
  u16*   Clo  = (u16*)carve(M * D * 2);

  // 1) split fp32 operands to bf16 hi/lo pairs
  {
    int n = (int)(M * D);
    split_kernel<<<(n + 255) / 256, 256, 0, stream>>>(X, Xhi, Xlo, n);
  }
  {
    int n = (int)(D * N3);
    split_kernel<<<(n + 255) / 256, 256, 0, stream>>>(Wqkv, W1hi, W1lo, n);
  }
  {
    int n = (int)(D * D);
    split_kernel<<<(n + 255) / 256, 256, 0, stream>>>(Wp, W2hi, W2lo, n);
  }

  // 2) fused QKV projection: QKV = X @ Wqkv + bqkv
  gemm_bf16x3<<<dim3((unsigned)(N3 / 128), (unsigned)(M / 128)), 256, 0, stream>>>(
      Xhi, Xlo, W1hi, W1lo, bqkv, QKV, (int)M, (int)N3, (int)D);

  // 3) per-head sum of V (for the eps/n numerator term)
  sumv_kernel<<<64, 64, 0, stream>>>(QKV, sumV);

  // 4) causal relu-softmax attention -> ctx (bf16 hi/lo)
  attn_kernel<<<dim3(64, 8), 256, 0, stream>>>(QKV, sumV, Chi, Clo);

  // 5) output projection: out = ctx @ Wp + bp
  gemm_bf16x3<<<dim3((unsigned)(D / 128), (unsigned)(M / 128)), 256, 0, stream>>>(
      Chi, Clo, W2hi, W2lo, bp, out, (int)M, (int)D, (int)D);
}
